// GRACE_84018150244509
// MI455X (gfx1250) — matrix operand order repaired and checked by reading, not compiled, not run on hardware
//
#include <hip/hip_runtime.h>

#define NN 8192
#define DD 128
#define JCHUNKS 4                    // j-dimension split for occupancy
#define JT_PER_CHUNK (NN / 16 / JCHUNKS)

typedef __bf16 bf16_t;
typedef __attribute__((ext_vector_type(16))) __bf16 v16bf;
typedef __attribute__((ext_vector_type(8)))  __bf16 v8bf;
typedef __attribute__((ext_vector_type(8)))  float  v8f;

// ---------------------------------------------------------------------------
// Kernel 0: split fp32 -> (hi, lo) bf16 pair.  hi = RNE(x), lo = RNE(x - hi).
// hi*hi + hi*lo + lo*hi recovers ~fp32 dot-product accuracy on bf16 WMMA.
// ---------------------------------------------------------------------------
__global__ void cvt_hilo(const float* __restrict__ z,
                         bf16_t* __restrict__ h,
                         bf16_t* __restrict__ l, int n) {
  int i = blockIdx.x * blockDim.x + threadIdx.x;
  if (i < n) {
    float x  = z[i];
    __bf16 hi = (__bf16)x;
    h[i] = hi;
    l[i] = (__bf16)(x - (float)hi);
  }
}

// A operand loader, contiguous k order (same as B, not the ISA table order):
// lane (m, half): VGPR0..7 hold K = 16*half + {0..15} contiguous (matches B)
__device__ __forceinline__ v16bf load_a(const bf16_t* __restrict__ row, int kb, int half) {
  v8bf c1 = *(const v8bf*)(row + kb + 16 * half);
  v8bf c2 = *(const v8bf*)(row + kb + 16 * half + 8);
  return __builtin_shufflevector(c1, c2, 0, 1, 2, 3, 4, 5, 6, 7,
                                         8, 9, 10, 11, 12, 13, 14, 15);
}

// ---------------------------------------------------------------------------
// Kernel 1: tiled similarity row-sums via v_wmma_f32_16x16x32_bf16.
// grid = (NN/128, 3, JCHUNKS): product 0: z1*z1^T, 1: z1*z2^T, 2: z2*z1^T.
// Each wave owns one 16-row tile and one exclusive j-chunk of 128 16-col
// tiles, accumulating sum_j exp(2*dot) per row into its own partial slot.
// Fully deterministic (no atomics; fixed combine order downstream).
// ---------------------------------------------------------------------------
__global__ __launch_bounds__(256) void sim_rowsums(
    const bf16_t* __restrict__ zh1, const bf16_t* __restrict__ zl1,
    const bf16_t* __restrict__ zh2, const bf16_t* __restrict__ zl2,
    float* __restrict__ R /* [JCHUNKS][3][NN] */) {
  const int lane = threadIdx.x & 31;
  const int wave = threadIdx.x >> 5;
  const int prod = blockIdx.y;
  const int chunk = blockIdx.z;
  const int rowBase = (blockIdx.x * 8 + wave) * 16;
  const int m    = lane & 15;
  const int half = lane >> 4;

  const bf16_t *Ah, *Al, *Bh, *Bl;
  if (prod == 0)      { Ah = zh1; Al = zl1; Bh = zh1; Bl = zl1; }
  else if (prod == 1) { Ah = zh1; Al = zl1; Bh = zh2; Bl = zl2; }
  else                { Ah = zh2; Al = zl2; Bh = zh1; Bl = zl1; }

  // Preload this wave's A rows (16 rows x K=128, hi+lo) into registers.
  const bf16_t* arH = Ah + (size_t)(rowBase + m) * DD;
  const bf16_t* arL = Al + (size_t)(rowBase + m) * DD;
  v16bf aH[4], aL[4];
#pragma unroll
  for (int kt = 0; kt < 4; ++kt) {
    aH[kt] = load_a(arH, kt * 32, half);
    aL[kt] = load_a(arL, kt * 32, half);
  }

  float part[8];
#pragma unroll
  for (int r = 0; r < 8; ++r) part[r] = 0.f;

  const int jt0 = chunk * JT_PER_CHUNK;
  for (int jt = jt0; jt < jt0 + JT_PER_CHUNK; ++jt) {
    // B operand: lane holds column n = m; K = 16*half + e contiguous.
    const bf16_t* brH = Bh + (size_t)(jt * 16 + m) * DD + 16 * half;
    const bf16_t* brL = Bl + (size_t)(jt * 16 + m) * DD + 16 * half;
    v8f acc = {};
#pragma unroll
    for (int kt = 0; kt < 4; ++kt) {
      v16bf bH = *(const v16bf*)(brH + kt * 32);
      v16bf bL = *(const v16bf*)(brL + kt * 32);
      acc = __builtin_amdgcn_wmma_f32_16x16x32_bf16(false, aH[kt], false, bH,
                                                    (short)0, acc, false, false);
      acc = __builtin_amdgcn_wmma_f32_16x16x32_bf16(false, aH[kt], false, bL,
                                                    (short)0, acc, false, false);
      acc = __builtin_amdgcn_wmma_f32_16x16x32_bf16(false, aL[kt], false, bH,
                                                    (short)0, acc, false, false);
    }
    // C layout: VGPR r = row (r + 8*half) at column N = m.  exp(dot/tau), tau=0.5.
#pragma unroll
    for (int r = 0; r < 8; ++r) part[r] += __expf(2.0f * acc[r]);
  }

  // Row sum = reduce across the 16 lanes of each half (xor masks stay in-half).
#pragma unroll
  for (int r = 0; r < 8; ++r) {
    float v = part[r];
    v += __shfl_xor(v, 1);
    v += __shfl_xor(v, 2);
    v += __shfl_xor(v, 4);
    v += __shfl_xor(v, 8);
    part[r] = v;
  }
  if (m == 0) {
    float* dst = R + ((size_t)chunk * 3 + prod) * NN + rowBase + half * 8;
#pragma unroll
    for (int r = 0; r < 8; ++r) dst[r] = part[r];
  }
}

// ---------------------------------------------------------------------------
// Kernel 2: per-row loss with exact fp32 diagonals + deterministic block sum.
// Combines the JCHUNKS partial row sums in a fixed order.
// ---------------------------------------------------------------------------
__global__ __launch_bounds__(256) void row_loss(
    const float* __restrict__ z1, const float* __restrict__ z2,
    const float* __restrict__ R, float* __restrict__ partial) {
  __shared__ float s[256];
  int i = blockIdx.x * 256 + threadIdx.x;
  const float* p1 = z1 + (size_t)i * DD;
  const float* p2 = z2 + (size_t)i * DD;
  float d11 = 0.f, d12 = 0.f;
#pragma unroll 8
  for (int k = 0; k < DD; ++k) {
    float a = p1[k];
    d11 = fmaf(a, a, d11);
    d12 = fmaf(a, p2[k], d12);
  }
  float dr = __expf(2.0f * d11);     // diag of refl
  float sp = 2.0f * d12;             // log(pos)

  float r1 = 0.f, r2 = 0.f, r3 = 0.f;
#pragma unroll
  for (int c = 0; c < JCHUNKS; ++c) {
    const float* Rc = R + (size_t)c * 3 * NN;
    r1 += Rc[i];
    r2 += Rc[NN + i];
    r3 += Rc[2 * NN + i];
  }
  float li = 0.5f * (__logf(r1 - dr + r2) + __logf(r1 - dr + r3)) - sp;

  s[threadIdx.x] = li;
  __syncthreads();
  for (int off = 128; off > 0; off >>= 1) {
    if ((int)threadIdx.x < off) s[threadIdx.x] += s[threadIdx.x + off];
    __syncthreads();
  }
  if (threadIdx.x == 0) partial[blockIdx.x] = s[0];
}

__global__ void finalize(const float* __restrict__ partial, float* __restrict__ out) {
  float s = 0.f;
  for (int b = 0; b < NN / 256; ++b) s += partial[b];
  *out = s / (float)NN;
}

// ---------------------------------------------------------------------------
extern "C" void kernel_launch(void* const* d_in, const int* in_sizes, int n_in,
                              void* d_out, int out_size, void* d_ws, size_t ws_size,
                              hipStream_t stream) {
  const float* z1 = (const float*)d_in[0];
  const float* z2 = (const float*)d_in[1];
  char* ws = (char*)d_ws;
  const size_t ND = (size_t)NN * DD;  // 1,048,576 elements

  // Workspace layout (bytes): 4 bf16 arrays of ND (2*ND B each) = 8*ND,
  // then R[JCHUNKS*3*NN] floats, then partial[NN/256] floats.  ~8.8 MB total.
  bf16_t* zh1 = (bf16_t*)(ws + 0 * ND);
  bf16_t* zl1 = (bf16_t*)(ws + 2 * ND);
  bf16_t* zh2 = (bf16_t*)(ws + 4 * ND);
  bf16_t* zl2 = (bf16_t*)(ws + 6 * ND);
  float*  R   = (float*)(ws + 8 * ND);
  float*  prt = (float*)(ws + 8 * ND + (size_t)JCHUNKS * 3 * NN * sizeof(float));

  const int thr = 256;
  const int cvtBlocks = (int)((ND + thr - 1) / thr);
  cvt_hilo<<<cvtBlocks, thr, 0, stream>>>(z1, zh1, zl1, (int)ND);
  cvt_hilo<<<cvtBlocks, thr, 0, stream>>>(z2, zh2, zl2, (int)ND);
  sim_rowsums<<<dim3(NN / 128, 3, JCHUNKS), 256, 0, stream>>>(zh1, zl1, zh2, zl2, R);
  row_loss<<<NN / 256, 256, 0, stream>>>(z1, z2, R, prt);
  finalize<<<1, 1, 0, stream>>>(prt, (float*)d_out);
}
